// Generator_74028056314209
// MI455X (gfx1250) — matrix operand order repaired and checked by reading, not compiled, not run on hardware
//
#include <hip/hip_runtime.h>
#include <cstdint>

typedef __attribute__((ext_vector_type(16))) _Float16 v16h;
typedef __attribute__((ext_vector_type(8)))  float    v8f;
typedef __attribute__((ext_vector_type(4)))  float    f4;

#define BM 128
#define BN 64
#define BK 32
#define LDP (BK + 4)   // LDS row stride in halves: 72B -> 8B-aligned packed stores

// ---------------------------------------------------------------------------
// Batched/strided GEMM:  C[z] = act( A[z] @ B[z] (+bias) (+resid) (+C) )
//   A: [M,K] row-major (lda). B: [K,N] row-major (ldb), or TRANSB: (k,n)=B[n*ldb+k]
//   Requires K % 4 == 0 (true for all uses here: 512, 100, 3072).
//   f32 global -> clamped unconditional b128 loads -> f16 LDS tiles
//   -> v_wmma_f32_16x16x32_f16 (4 per wave per k-step) -> f32 epilogue.
// 256 threads = 8 wave32 in a 4x2 grid; each wave owns a 32x32 output tile.
// ---------------------------------------------------------------------------
template <int TRANSB>
__launch_bounds__(256)
__global__ void gemm_wmma_f16(
    int M, int N, int K,
    const float* __restrict__ A, int lda, long long aBs,
    const float* __restrict__ B, int ldb, long long bBs,
    float* __restrict__ C, int ldc, long long cBs,
    const float* __restrict__ bias,
    const float* __restrict__ resid, int ldr, long long rBs,
    int accum, int act)
{
  __shared__ _Float16 As[BM][LDP];
  __shared__ _Float16 Bs[BN][LDP];

  const int z = blockIdx.z;
  A += (size_t)z * aBs;
  B += (size_t)z * bBs;
  C += (size_t)z * cBs;
  if (resid) resid += (size_t)z * rBs;

  const int tid  = threadIdx.x;
  const int lane = tid & 31;
  const int wid  = tid >> 5;
  const int m0   = blockIdx.y * BM;
  const int n0   = blockIdx.x * BN;
  const int wm   = (wid & 3) << 5;   // wave 32-row strip
  const int wn   = (wid >> 2) << 5;  // wave 32-col strip

  v8f acc00 = {}, acc01 = {}, acc10 = {}, acc11 = {};

  // ---- register stage for the current k-tile ----
  f4 sa[4];   // A: 4 float4 per thread (128x32)
  f4 sb[2];   // B: 2 float4 per thread (32x64)
  const f4 fzero = {0.f, 0.f, 0.f, 0.f};

  // Per-thread invariant indices for the staging pattern
  const int ar  = (tid >> 3) >> 0;          // unused placeholder (kept simple below)
  (void)ar;

  auto load_stage = [&](int kb) {
    // A tile: slot e in [0,1024): r=e/8, c4=(e%8)*4
#pragma unroll
    for (int p = 0; p < 4; ++p) {
      int e  = tid + p * 256;
      int r  = e >> 3;
      int c4 = (e & 7) << 2;
      int gr = m0 + r;
      int gk = kb + c4;
      int grc = gr < M ? gr : M - 1;
      bool kok = (gk + 4 <= K);
      int gkc = kok ? gk : 0;
      f4 v = *(const f4*)&A[(size_t)grc * lda + gkc];
      if (!((gr < M) & kok)) v = fzero;
      sa[p] = v;
    }
    if (TRANSB) {
      // slot e in [0,512): nn=e/8, kk4=(e%8)*4 ; contiguous along K
#pragma unroll
      for (int p = 0; p < 2; ++p) {
        int e   = tid + p * 256;
        int nn  = e >> 3;
        int kk4 = (e & 7) << 2;
        int gn  = n0 + nn;
        int gk  = kb + kk4;
        int gnc = gn < N ? gn : N - 1;
        bool kok = (gk + 4 <= K);
        int gkc = kok ? gk : 0;
        f4 v = *(const f4*)&B[(size_t)gnc * ldb + gkc];
        if (!((gn < N) & kok)) v = fzero;
        sb[p] = v;
      }
    } else {
      // slot e in [0,512): kk=e/16, nn4=(e%16)*4 ; contiguous along N
#pragma unroll
      for (int p = 0; p < 2; ++p) {
        int e   = tid + p * 256;
        int kk  = e >> 4;
        int nn4 = (e & 15) << 2;
        int gk  = kb + kk;
        int gn  = n0 + nn4;
        int gkc = gk < K ? gk : K - 1;
        f4 v;
        if ((gn + 4 <= N) & (gk < K)) {
          v = *(const f4*)&B[(size_t)gkc * ldb + gn];
        } else {
          v = fzero;
#pragma unroll
          for (int j = 0; j < 4; ++j) {
            int gnj = gn + j;
            int gnc = gnj < N ? gnj : N - 1;
            float s = B[(size_t)gkc * ldb + gnc];
            v[j] = ((gk < K) & (gnj < N)) ? s : 0.f;
          }
        }
        sb[p] = v;
      }
    }
  };

  load_stage(0);

  for (int kb = 0; kb < K; kb += BK) {
    // ---- commit stage to LDS (f32 -> f16, packed 8B stores) ----
#pragma unroll
    for (int p = 0; p < 4; ++p) {
      int e  = tid + p * 256;
      int r  = e >> 3;
      int c4 = (e & 7) << 2;
      union { _Float16 h[4]; unsigned long long u; } pk;
#pragma unroll
      for (int j = 0; j < 4; ++j) pk.h[j] = (_Float16)sa[p][j];
      *(unsigned long long*)&As[r][c4] = pk.u;
    }
    if (TRANSB) {
#pragma unroll
      for (int p = 0; p < 2; ++p) {
        int e   = tid + p * 256;
        int nn  = e >> 3;
        int kk4 = (e & 7) << 2;
        union { _Float16 h[4]; unsigned long long u; } pk;
#pragma unroll
        for (int j = 0; j < 4; ++j) pk.h[j] = (_Float16)sb[p][j];
        *(unsigned long long*)&Bs[nn][kk4] = pk.u;
      }
    } else {
#pragma unroll
      for (int p = 0; p < 2; ++p) {
        int e   = tid + p * 256;
        int kk  = e >> 4;
        int nn4 = (e & 15) << 2;
#pragma unroll
        for (int j = 0; j < 4; ++j) Bs[nn4 + j][kk] = (_Float16)sb[p][j];
      }
    }
    __syncthreads();

    // ---- prefetch next k-tile into registers (overlaps with WMMA below) ----
    if (kb + BK < K) load_stage(kb + BK);

    // ---- fragments in contiguous k order: lane half h holds k = 16h..16h+15 ----
    union Frag { v16h v; unsigned u[8]; };
    Frag fa0, fa1, fb0, fb1;
    const int rl    = lane & 15;
    const int khalf = (lane >> 4) << 4;   // A: lanes 16-31 are K 16..31
    const int kb2   = (lane >> 4) << 4;   // B: lanes 16-31 are K 16..31
#pragma unroll
    for (int j = 0; j < 8; ++j) {
      int k0 = khalf + (j << 1);
      fa0.u[j] = *(const unsigned*)&As[wm + rl][k0];
      fa1.u[j] = *(const unsigned*)&As[wm + 16 + rl][k0];
    }
#pragma unroll
    for (int j = 0; j < 8; ++j) {
      fb0.u[j] = *(const unsigned*)&Bs[wn + rl][kb2 + 2 * j];
      fb1.u[j] = *(const unsigned*)&Bs[wn + 16 + rl][kb2 + 2 * j];
    }

    acc00 = __builtin_amdgcn_wmma_f32_16x16x32_f16(false, fa0.v, false, fb0.v, (short)0, acc00, false, false);
    acc01 = __builtin_amdgcn_wmma_f32_16x16x32_f16(false, fa0.v, false, fb1.v, (short)0, acc01, false, false);
    acc10 = __builtin_amdgcn_wmma_f32_16x16x32_f16(false, fa1.v, false, fb0.v, (short)0, acc10, false, false);
    acc11 = __builtin_amdgcn_wmma_f32_16x16x32_f16(false, fa1.v, false, fb1.v, (short)0, acc11, false, false);
    __syncthreads();
  }

  // ---- epilogue: C/D layout lane = N%16, rows r + 8*(lane>>4) ----
  const int rbase = m0 + wm + ((lane >> 4) << 3);
  const int cbase = n0 + wn + (lane & 15);
#pragma unroll
  for (int i = 0; i < 2; ++i) {
#pragma unroll
    for (int jt = 0; jt < 2; ++jt) {
      const v8f* ac = i == 0 ? (jt == 0 ? &acc00 : &acc01)
                             : (jt == 0 ? &acc10 : &acc11);
      int gc = cbase + jt * 16;
      if (gc >= N) continue;
#pragma unroll
      for (int r = 0; r < 8; ++r) {
        int gr = rbase + i * 16 + r;
        if (gr >= M) continue;
        float v = (*ac)[r];
        size_t ci = (size_t)gr * ldc + gc;
        if (bias)  v += bias[gc];
        if (resid) v += resid[(size_t)gr * ldr + gc];
        if (accum) v += C[ci];
        if (act == 1)      v = (v >= 0.f) ? v : 0.2f * v;   // leaky
        else if (act == 2) v = (__sinf(v) + 1.f) * 0.5f;     // sine_one
        C[ci] = v;
      }
    }
  }
}

// enc[b,s,c] = members[s,c] * w[b,c] + bias[b,c]
__global__ void setexp_kernel(const float* __restrict__ members,
                              const float* __restrict__ w,
                              const float* __restrict__ b,
                              float* __restrict__ enc)
{
  int i = blockIdx.x * blockDim.x + threadIdx.x;
  if (i >= 64 * 100 * 512) return;
  int c  = i & 511;
  int s  = (i >> 9) % 100;
  int bb = i / (100 * 512);
  enc[i] = members[s * 512 + c] * w[bb * 512 + c] + b[bb * 512 + c];
}

// softmax over last dim (100 cols), one wave32 per row, shuffle reductions
__global__ void softmax_rows100(float* __restrict__ S, int nrows)
{
  int row  = blockIdx.x * 8 + (threadIdx.x >> 5);
  int lane = threadIdx.x & 31;
  if (row >= nrows) return;
  float* p = S + (size_t)row * 100;
  float v[4];
  float m = -3.4e38f;
#pragma unroll
  for (int i = 0; i < 4; ++i) {
    int c = lane + i * 32;
    v[i] = (c < 100) ? p[c] : -3.4e38f;
    m = fmaxf(m, v[i]);
  }
  for (int off = 16; off; off >>= 1) m = fmaxf(m, __shfl_xor(m, off, 32));
  float s = 0.f;
#pragma unroll
  for (int i = 0; i < 4; ++i) {
    int c = lane + i * 32;
    if (c < 100) { v[i] = __expf(v[i] - m); s += v[i]; }
  }
  for (int off = 16; off; off >>= 1) s += __shfl_xor(s, off, 32);
  float inv = 1.f / s;
#pragma unroll
  for (int i = 0; i < 4; ++i) {
    int c = lane + i * 32;
    if (c < 100) p[c] = v[i] * inv;
  }
}

// softmax over S (dim=1) of logits [64,100,3072]: one thread per (b, n)
__global__ void softmax_cols_S(float* __restrict__ L)
{
  int i = blockIdx.x * blockDim.x + threadIdx.x;
  if (i >= 64 * 3072) return;
  int b = i / 3072, n = i % 3072;
  float* p = L + (size_t)b * 100 * 3072 + n;
  float m = -3.4e38f;
  for (int s = 0; s < 100; ++s) m = fmaxf(m, p[(size_t)s * 3072]);
  float sum = 0.f;
  for (int s = 0; s < 100; ++s) sum += __expf(p[(size_t)s * 3072] - m);
  float inv = 1.f / sum;
  for (int s = 0; s < 100; ++s)
    p[(size_t)s * 3072] = __expf(p[(size_t)s * 3072] - m) * inv;
}

// ---------------------------------------------------------------------------
// Host side
// ---------------------------------------------------------------------------
static inline void gemm(hipStream_t st, int M, int N, int K,
                        const float* A, int lda, long long aBs,
                        const float* B, int ldb, long long bBs, int tB,
                        float* C, int ldc, long long cBs,
                        const float* bias, const float* resid, int ldr,
                        long long rBs, int accum, int act, int batch)
{
  dim3 g((N + BN - 1) / BN, (M + BM - 1) / BM, batch), blk(256);
  if (tB)
    gemm_wmma_f16<1><<<g, blk, 0, st>>>(M, N, K, A, lda, aBs, B, ldb, bBs,
                                        C, ldc, cBs, bias, resid, ldr, rBs, accum, act);
  else
    gemm_wmma_f16<0><<<g, blk, 0, st>>>(M, N, K, A, lda, aBs, B, ldb, bBs,
                                        C, ldc, cBs, bias, resid, ldr, rBs, accum, act);
}

struct Lay {
  const float *Wq, *bq, *Wk, *bk, *Wv, *bv, *fcw, *fcb;
  const float *sw0, *sb0, *sw1, *sb1, *sw2, *sb2;
};

extern "C" void kernel_launch(void* const* d_in, const int* in_sizes, int n_in,
                              void* d_out, int out_size, void* d_ws, size_t ws_size,
                              hipStream_t stream)
{
  (void)out_size; (void)ws_size;
  const float* P[200];
  int np = n_in < 200 ? n_in : 200;
  for (int i = 0; i < np; ++i) P[i] = (const float*)d_in[i];

  const float *x, *members, *emb;
  const float *sbw[4], *sbb[4], *sww[4], *swb[4];
  const float *atw[4], *atb[4], *pw[4], *pb[4];
  Lay L[8];

  // Two candidate pytree flatten orders; detect from leading sizes.
  bool ins = (n_in >= 147) && (in_sizes[0] == 64 * 512) && (in_sizes[1] == 100 * 512);
  if (ins) {
    // dict insertion order: x, members, se_bias, se_weight, layers, atoms, pos, emb
    x = P[0]; members = P[1];
    for (int i = 0; i < 4; ++i) { sbw[i] = P[2 + 2 * i];  sbb[i] = P[3 + 2 * i]; }
    for (int i = 0; i < 4; ++i) { sww[i] = P[10 + 2 * i]; swb[i] = P[11 + 2 * i]; }
    for (int li = 0; li < 8; ++li) {
      int b = 18 + li * 14;
      L[li].Wq = P[b];     L[li].bq = P[b + 1];
      L[li].Wk = P[b + 2]; L[li].bk = P[b + 3];
      L[li].Wv = P[b + 4]; L[li].bv = P[b + 5];
      L[li].fcw = P[b + 6]; L[li].fcb = P[b + 7];
      L[li].sw0 = P[b + 8];  L[li].sb0 = P[b + 9];
      L[li].sw1 = P[b + 10]; L[li].sb1 = P[b + 11];
      L[li].sw2 = P[b + 12]; L[li].sb2 = P[b + 13];
    }
    for (int i = 0; i < 4; ++i) { atw[i] = P[130 + 2 * i]; atb[i] = P[131 + 2 * i]; }
    for (int i = 0; i < 4; ++i) { pw[i]  = P[138 + 2 * i]; pb[i]  = P[139 + 2 * i]; }
    emb = P[146];
  } else {
    // jax sorted-key order: params{atoms, attn_layers(Wk,Wq,Wv,bk,bq,bv,fcb,fcw,stack),
    //                              emb, members, pos, se_bias, se_weight}, x
    for (int i = 0; i < 4; ++i) { atw[i] = P[0 + 2 * i]; atb[i] = P[1 + 2 * i]; }
    for (int li = 0; li < 8; ++li) {
      int b = 8 + li * 14;
      L[li].Wk = P[b];     L[li].Wq = P[b + 1]; L[li].Wv = P[b + 2];
      L[li].bk = P[b + 3]; L[li].bq = P[b + 4]; L[li].bv = P[b + 5];
      L[li].fcb = P[b + 6]; L[li].fcw = P[b + 7];
      L[li].sw0 = P[b + 8];  L[li].sb0 = P[b + 9];
      L[li].sw1 = P[b + 10]; L[li].sb1 = P[b + 11];
      L[li].sw2 = P[b + 12]; L[li].sb2 = P[b + 13];
    }
    emb = P[120]; members = P[121];
    for (int i = 0; i < 4; ++i) { pw[i]  = P[122 + 2 * i]; pb[i]  = P[123 + 2 * i]; }
    for (int i = 0; i < 4; ++i) { sbw[i] = P[130 + 2 * i]; sbb[i] = P[131 + 2 * i]; }
    for (int i = 0; i < 4; ++i) { sww[i] = P[138 + 2 * i]; swb[i] = P[139 + 2 * i]; }
    x = P[146];
  }

  // Workspace layout (floats)
  float* W = (float*)d_ws;
  const size_t RC = 6400ull * 512;
  float* t0  = W; W += RC;
  float* t1  = W; W += RC;
  float* enc = W; W += RC;
  float* en2 = W; W += RC;
  float* qh  = W; W += RC;
  float* kh  = W; W += RC;
  float* vh  = W; W += RC;
  float* oh  = W; W += RC;
  float* sc  = W; W += 64ull * 100 * 100;
  float* logits = W;                      // 6400*3072

  float* out = (float*)d_out;             // [6400,130]

  // --- SetExpansion: bias stack (final -> t1), weight stack (final -> qh) ---
  gemm(stream, 64, 512, 512, x,  512, 0, sbw[0], 512, 0, 0, t0, 512, 0, sbb[0], nullptr, 0, 0, 0, 1, 1);
  gemm(stream, 64, 512, 512, t0, 512, 0, sbw[1], 512, 0, 0, t1, 512, 0, sbb[1], nullptr, 0, 0, 0, 1, 1);
  gemm(stream, 64, 512, 512, t1, 512, 0, sbw[2], 512, 0, 0, t0, 512, 0, sbb[2], nullptr, 0, 0, 0, 1, 1);
  gemm(stream, 64, 512, 512, t0, 512, 0, sbw[3], 512, 0, 0, t1, 512, 0, sbb[3], nullptr, 0, 0, 0, 0, 1);
  gemm(stream, 64, 512, 512, x,  512, 0, sww[0], 512, 0, 0, t0, 512, 0, swb[0], nullptr, 0, 0, 0, 1, 1);
  gemm(stream, 64, 512, 512, t0, 512, 0, sww[1], 512, 0, 0, qh, 512, 0, swb[1], nullptr, 0, 0, 0, 1, 1);
  gemm(stream, 64, 512, 512, qh, 512, 0, sww[2], 512, 0, 0, t0, 512, 0, swb[2], nullptr, 0, 0, 0, 1, 1);
  gemm(stream, 64, 512, 512, t0, 512, 0, sww[3], 512, 0, 0, qh, 512, 0, swb[3], nullptr, 0, 0, 0, 0, 1);
  setexp_kernel<<<(64 * 100 * 512) / 256, 256, 0, stream>>>(members, qh, t1, enc);

  // --- Attention stack: 8 layers, per-head loop, fc accumulated into en2 ---
  for (int li = 0; li < 8; ++li) {
    const Lay& l = L[li];
    for (int h = 0; h < 8; ++h) {
      const long long HW = 512ll * 512;
      gemm(stream, 6400, 512, 512, enc, 512, 0, l.Wq + (size_t)h * HW, 512, 0, 0,
           qh, 512, 0, l.bq + h * 512, nullptr, 0, 0, 0, 0, 1);
      gemm(stream, 6400, 512, 512, enc, 512, 0, l.Wk + (size_t)h * HW, 512, 0, 0,
           kh, 512, 0, l.bk + h * 512, nullptr, 0, 0, 0, 0, 1);
      gemm(stream, 6400, 512, 512, enc, 512, 0, l.Wv + (size_t)h * HW, 512, 0, 0,
           vh, 512, 0, l.bv + h * 512, nullptr, 0, 0, 0, 0, 1);
      // scores[b] = q[b] @ k[b]^T   (unscaled, as in source)
      gemm(stream, 100, 100, 512, qh, 512, 51200, kh, 512, 51200, 1,
           sc, 100, 10000, nullptr, nullptr, 0, 0, 0, 0, 64);
      softmax_rows100<<<800, 256, 0, stream>>>(sc, 6400);
      // o[b] = attn[b] @ v[b]
      gemm(stream, 100, 512, 100, sc, 100, 10000, vh, 512, 51200, 0,
           oh, 512, 51200, nullptr, nullptr, 0, 0, 0, 0, 64);
      // en2 (+)= o_h @ fcw[h*512:(h+1)*512, :]; h==0 adds residual enc + fcb
      if (h == 0)
        gemm(stream, 6400, 512, 512, oh, 512, 0, l.fcw + (size_t)h * HW, 512, 0, 0,
             en2, 512, 0, l.fcb, enc, 512, 0, 0, 0, 1);
      else
        gemm(stream, 6400, 512, 512, oh, 512, 0, l.fcw + (size_t)h * HW, 512, 0, 0,
             en2, 512, 0, nullptr, nullptr, 0, 0, 1, 0, 1);
    }
    // per-layer LinearOutputStack (2 hidden + final) -> enc
    gemm(stream, 6400, 512, 512, en2, 512, 0, l.sw0, 512, 0, 0, t0, 512, 0, l.sb0, nullptr, 0, 0, 0, 1, 1);
    gemm(stream, 6400, 512, 512, t0,  512, 0, l.sw1, 512, 0, 0, t1, 512, 0, l.sb1, nullptr, 0, 0, 0, 1, 1);
    gemm(stream, 6400, 512, 512, t1,  512, 0, l.sw2, 512, 0, 0, enc, 512, 0, l.sb2, nullptr, 0, 0, 0, 0, 1);
  }

  // --- atoms head: logits, softmax over S, @emb into out[:, :128] ---
  gemm(stream, 6400, 512, 512,  enc, 512, 0, atw[0], 512, 0, 0,  t0, 512, 0, atb[0], nullptr, 0, 0, 0, 1, 1);
  gemm(stream, 6400, 512, 512,  t0,  512, 0, atw[1], 512, 0, 0,  t1, 512, 0, atb[1], nullptr, 0, 0, 0, 1, 1);
  gemm(stream, 6400, 512, 512,  t1,  512, 0, atw[2], 512, 0, 0,  t0, 512, 0, atb[2], nullptr, 0, 0, 0, 1, 1);
  gemm(stream, 6400, 3072, 512, t0,  512, 0, atw[3], 3072, 0, 0, logits, 3072, 0, atb[3], nullptr, 0, 0, 0, 0, 1);
  softmax_cols_S<<<(64 * 3072 + 255) / 256, 256, 0, stream>>>(logits);
  gemm(stream, 6400, 128, 3072, logits, 3072, 0, emb, 128, 0, 0,
       out, 130, 0, nullptr, nullptr, 0, 0, 0, 0, 1);

  // --- pos head: sine_one into out[:, 128:130] ---
  gemm(stream, 6400, 512, 512, enc, 512, 0, pw[0], 512, 0, 0, t1, 512, 0, pb[0], nullptr, 0, 0, 0, 1, 1);
  gemm(stream, 6400, 512, 512, t1,  512, 0, pw[1], 512, 0, 0, t0, 512, 0, pb[1], nullptr, 0, 0, 0, 1, 1);
  gemm(stream, 6400, 512, 512, t0,  512, 0, pw[2], 512, 0, 0, t1, 512, 0, pb[2], nullptr, 0, 0, 0, 1, 1);
  gemm(stream, 6400, 2, 512,   t1,  512, 0, pw[3], 2, 0, 0, out + 128, 130, 0, pb[3], nullptr, 0, 0, 0, 2, 1);
}
